// GraphAttentionLayerV2_12532714570497
// MI455X (gfx1250) — matrix operand order repaired and checked by reading, not compiled, not run on hardware
//
#include <hip/hip_runtime.h>
#include <math.h>

// ---------------- problem constants ----------------
#define BB   4
#define SS   512
#define HIDN 768
#define NH   12
#define HD   64
#define FFD  3072
#define RELD 64
#define BS   (BB*SS)          // 2048 rows of hidden states

typedef __bf16 bf16_t;
typedef __attribute__((ext_vector_type(16))) __bf16 v16bf;
typedef __attribute__((ext_vector_type(8)))  __bf16 v8bf;
typedef __attribute__((ext_vector_type(8)))  float  v8f;
typedef __attribute__((ext_vector_type(4)))  float  v4f;

__device__ __forceinline__ bf16_t f2bf(float f) {
    unsigned u = __builtin_bit_cast(unsigned, f);
    unsigned r = u + 0x7fffu + ((u >> 16) & 1u);   // round-to-nearest-even
    unsigned short h = (unsigned short)(r >> 16);
    return __builtin_bit_cast(bf16_t, h);
}
__device__ __forceinline__ bf16_t bfzero() {
    unsigned short z = 0; return __builtin_bit_cast(bf16_t, z);
}

// ----- fragment loaders (unguarded fast path) -----
// A (16x32 bf16, MxK): lane l<16 holds row m0+l, K {0..7, 16..23}; lane l+16: K {8..15, 24..31}.
// caller passes p = rowbase + k0 + half*8
__device__ __forceinline__ v16bf ldA(const bf16_t* p) {
    v8bf a = *(const v8bf*)p;
    v8bf b = *(const v8bf*)(p + 16);
    v16bf r;
#pragma unroll
    for (int i = 0; i < 8; i++) { r[i] = a[i]; r[i + 8] = b[i]; }
    return r;
}
__device__ __forceinline__ v16bf ldA(const float* p) {
    v4f a = *(const v4f*)p;
    v4f b = *(const v4f*)(p + 4);
    v4f c = *(const v4f*)(p + 16);
    v4f d = *(const v4f*)(p + 20);
    v16bf r;
#pragma unroll
    for (int i = 0; i < 4; i++) {
        r[i] = f2bf(a[i]); r[4 + i] = f2bf(b[i]);
        r[8 + i] = f2bf(c[i]); r[12 + i] = f2bf(d[i]);
    }
    return r;
}
template <typename AT>
__device__ __forceinline__ v16bf ldAg(const AT* p, bool ok) {   // guarded (M%16!=0 path)
    if (ok) return ldA(p);
    v16bf r; bf16_t z = bfzero();
#pragma unroll
    for (int i = 0; i < 16; i++) r[i] = z;
    return r;
}
// B (32x16 bf16, KxN), stored as Bt (NxK row-major): lane l<16 holds col n, K {0..7, 16..23};
// lane l+16 holds K {8..15, 24..31}. caller passes p = colbase + k0 + half*8
__device__ __forceinline__ v16bf ldB(const bf16_t* p) {
    v8bf a = *(const v8bf*)p;
    v8bf b = *(const v8bf*)(p + 16);
    v16bf r;
#pragma unroll
    for (int i = 0; i < 8; i++) { r[i] = a[i]; r[i + 8] = b[i]; }
    return r;
}

// ---------------- generic WMMA GEMM ----------------
// C[m,n] (+=) sum_k A[m,k] * Bt[n,k]
// EPI: 0=f32 out (+bias,+resid,+ACC), 1=bf16 out (+bias), 2=gelu->bf16 out (+bias)
// MG:  true only when M is not a multiple of 16 (adds row guards)
// N must be a multiple of 16*NT, K a multiple of 64 (ping-pong unrolled loop).
// Batched on blockIdx.y/z with independent element strides.
template <typename AT, int NT, int EPI, bool ACC, bool MG>
__global__ __launch_bounds__(256) void gemm_wmma(
    const AT* __restrict__ A, long sAy, long sAz, int lda,
    const bf16_t* __restrict__ Bt, long sBy, long sBz, int ldb,
    void* __restrict__ Cp, long sCy, long sCz, int ldc,
    int M, int N, int K,
    const float* __restrict__ bias, const float* __restrict__ resid)
{
    int wid  = (int)((blockIdx.x * blockDim.x + threadIdx.x) >> 5);
    int lane = threadIdx.x & 31;
    int half = lane >> 4;
    int l    = lane & 15;
    int ntiles = N / (16 * NT);
    int mtiles = (M + 15) >> 4;
    if (wid >= mtiles * ntiles) return;
    int m0 = (wid / ntiles) * 16;
    int n0 = (wid % ntiles) * (16 * NT);

    const AT*     Ab = A  + (long)blockIdx.y * sAy + (long)blockIdx.z * sAz;
    const bf16_t* Bb = Bt + (long)blockIdx.y * sBy + (long)blockIdx.z * sBz;
    long coff = (long)blockIdx.y * sCy + (long)blockIdx.z * sCz;

    v8f zero = {0.f, 0.f, 0.f, 0.f, 0.f, 0.f, 0.f, 0.f};
    v8f acc[NT];
#pragma unroll
    for (int t = 0; t < NT; t++) acc[t] = zero;

    int  arow = m0 + l;
    bool aok  = !MG || (arow < M);
    const AT* aptr = Ab + (long)arow * lda + half * 8;

    const bf16_t* bptr[NT];
#pragma unroll
    for (int t = 0; t < NT; t++)
        bptr[t] = Bb + (long)(n0 + t * 16 + l) * ldb + half * 8;

    // ping-pong pipelined K loop (K % 64 == 0): no cross-buffer register rotation
    v16bf a0 = MG ? ldAg(aptr, aok) : ldA(aptr);
    v16bf b0[NT];
#pragma unroll
    for (int t = 0; t < NT; t++) b0[t] = ldB(bptr[t]);

    int k0 = 0;
    for (; k0 + 64 < K; k0 += 64) {
        // fetch second half of this 64-chunk
        v16bf a1 = MG ? ldAg(aptr + k0 + 32, aok) : ldA(aptr + k0 + 32);
        v16bf b1[NT];
#pragma unroll
        for (int t = 0; t < NT; t++) b1[t] = ldB(bptr[t] + k0 + 32);
        // multiply first half (loads issued last iteration)
#pragma unroll
        for (int t = 0; t < NT; t++)
            acc[t] = __builtin_amdgcn_wmma_f32_16x16x32_bf16(
                false, a0, false, b0[t], (short)0, acc[t], false, false);
        // reload buffer 0 with next chunk (its old value is dead now)
        a0 = MG ? ldAg(aptr + k0 + 64, aok) : ldA(aptr + k0 + 64);
#pragma unroll
        for (int t = 0; t < NT; t++) b0[t] = ldB(bptr[t] + k0 + 64);
        // multiply second half
#pragma unroll
        for (int t = 0; t < NT; t++)
            acc[t] = __builtin_amdgcn_wmma_f32_16x16x32_bf16(
                false, a1, false, b1[t], (short)0, acc[t], false, false);
    }
    {   // tail: last 64-chunk (b0/a0 hold k0; fetch k0+32)
        v16bf a1 = MG ? ldAg(aptr + k0 + 32, aok) : ldA(aptr + k0 + 32);
        v16bf b1[NT];
#pragma unroll
        for (int t = 0; t < NT; t++) b1[t] = ldB(bptr[t] + k0 + 32);
#pragma unroll
        for (int t = 0; t < NT; t++)
            acc[t] = __builtin_amdgcn_wmma_f32_16x16x32_bf16(
                false, a0, false, b0[t], (short)0, acc[t], false, false);
#pragma unroll
        for (int t = 0; t < NT; t++)
            acc[t] = __builtin_amdgcn_wmma_f32_16x16x32_bf16(
                false, a1, false, b1[t], (short)0, acc[t], false, false);
    }

    // epilogue: VGPR v <-> row m0+v (lanes 0-15) / m0+v+8 (lanes 16-31); col = n0+t*16+l
#pragma unroll
    for (int t = 0; t < NT; t++) {
        int n = n0 + t * 16 + l;
#pragma unroll
        for (int v = 0; v < 8; v++) {
            int m = m0 + v + half * 8;
            if (MG && m >= M) continue;
            long idx = (long)m * ldc + n;
            float c = acc[t][v];
            if (bias) c += bias[n];
            if (EPI == 0) {
                float* Co = (float*)Cp + coff;
                if (resid) c += resid[idx];
                if (ACC)   c += Co[idx];
                Co[idx] = c;
            } else if (EPI == 1) {
                ((bf16_t*)Cp + coff)[idx] = f2bf(c);
            } else {
                float g = 0.5f * c * (1.f + erff(c * 0.7071067811865475f));
                ((bf16_t*)Cp + coff)[idx] = f2bf(g);
            }
        }
    }
}

// ---------------- small elementwise / reshape kernels ----------------
__global__ __launch_bounds__(256) void f2bf_kernel(const float* __restrict__ x,
                                                   bf16_t* __restrict__ y, long n) {
    long i = (long)blockIdx.x * 256 + threadIdx.x;
    if (i < n) y[i] = f2bf(x[i]);
}

// Wt[n*Kd + k] = bf16(W[k*Nd + n])   (W stored (in=K, out=N))
__global__ __launch_bounds__(256) void transpose_bf_kernel(const float* __restrict__ W,
                                                           bf16_t* __restrict__ Wt,
                                                           int Kd, int Nd) {
    long i = (long)blockIdx.x * 256 + threadIdx.x;
    if (i >= (long)Kd * Nd) return;
    int n = (int)(i / Kd), k = (int)(i % Kd);
    Wt[i] = f2bf(W[(long)k * Nd + n]);
}

__global__ __launch_bounds__(256) void pack_qkv_kernel(
    const float* __restrict__ qf, const float* __restrict__ kf, const float* __restrict__ vf,
    bf16_t* __restrict__ qb, bf16_t* __restrict__ kb, bf16_t* __restrict__ vtb) {
    long i = (long)blockIdx.x * 256 + threadIdx.x;
    if (i >= (long)BS * HIDN) return;
    int m = (int)(i / HIDN), n = (int)(i % HIDN);
    int b = m / SS, s = m % SS, h = n / HD, d = n % HD;
    long bh = (long)b * NH + h;
    qb[(bh * SS + s) * HD + d]  = f2bf(qf[i]);
    kb[(bh * SS + s) * HD + d]  = f2bf(kf[i]);
    vtb[(bh * HD + d) * SS + s] = f2bf(vf[i]);   // transposed: [b,h,d,s]
}

__global__ __launch_bounds__(256) void pack_ctx_kernel(const float* __restrict__ cf,
                                                       bf16_t* __restrict__ cb) {
    long i = (long)blockIdx.x * 256 + threadIdx.x;
    if (i >= (long)BB * NH * SS * HD) return;
    int bh = (int)(i / (SS * HD));
    int r  = (int)(i % (SS * HD));
    int s = r / HD, d = r % HD;
    int b = bh / NH, h = bh % NH;
    cb[((long)(b * SS + s)) * HIDN + h * HD + d] = f2bf(cf[i]);
}

// softmax over j: x = scores/8 + (1-graph)*(-1e9); grid = (S, H, B)
__global__ __launch_bounds__(256) void softmax_kernel(const float* __restrict__ scores,
                                                      const float* __restrict__ graph,
                                                      bf16_t* __restrict__ probs) {
    int i = blockIdx.x, h = blockIdx.y, b = blockIdx.z;
    long rowo = (((long)(b * NH + h)) * SS + i) * SS;
    const float* srow = scores + rowo;
    const float* grow = graph + ((long)(b * SS) + i) * SS;
    bf16_t* prow = probs + rowo;
    int t = threadIdx.x;
    __shared__ float red[256];
    float x0 = srow[t]       * 0.125f + (1.f - grow[t])       * (-1.0e9f);
    float x1 = srow[t + 256] * 0.125f + (1.f - grow[t + 256]) * (-1.0e9f);
    red[t] = fmaxf(x0, x1);
    __syncthreads();
    for (int s = 128; s > 0; s >>= 1) { if (t < s) red[t] = fmaxf(red[t], red[t + s]); __syncthreads(); }
    float mx = red[0];
    __syncthreads();
    float e0 = expf(x0 - mx), e1 = expf(x1 - mx);
    red[t] = e0 + e1;
    __syncthreads();
    for (int s = 128; s > 0; s >>= 1) { if (t < s) red[t] += red[t + s]; __syncthreads(); }
    float inv = 1.f / red[0];
    prow[t] = f2bf(e0 * inv);
    prow[t + 256] = f2bf(e1 * inv);
}

// layernorm over HIDN=768 (3 elems/thread); writes f32 and optional bf16 copy
__global__ __launch_bounds__(256) void ln_kernel(const float* __restrict__ x,
                                                 const float* __restrict__ g,
                                                 const float* __restrict__ bt,
                                                 float* __restrict__ outf,
                                                 bf16_t* __restrict__ outb) {
    long row = blockIdx.x;
    const float* xr = x + row * HIDN;
    int t = threadIdx.x;
    __shared__ float red[256];
    float v0 = xr[t], v1 = xr[t + 256], v2 = xr[t + 512];
    red[t] = v0 + v1 + v2;
    __syncthreads();
    for (int s = 128; s > 0; s >>= 1) { if (t < s) red[t] += red[t + s]; __syncthreads(); }
    float u = red[0] * (1.0f / HIDN);
    __syncthreads();
    float d0 = v0 - u, d1 = v1 - u, d2 = v2 - u;
    red[t] = d0 * d0 + d1 * d1 + d2 * d2;
    __syncthreads();
    for (int s = 128; s > 0; s >>= 1) { if (t < s) red[t] += red[t + s]; __syncthreads(); }
    float inv = rsqrtf(red[0] * (1.0f / HIDN) + 1e-12f);
    float dv[3] = {d0, d1, d2};
#pragma unroll
    for (int c = 0; c < 3; c++) {
        int j = t + c * 256;
        float val = g[j] * dv[c] * inv + bt[j];
        if (outf) outf[row * HIDN + j] = val;
        if (outb) outb[row * HIDN + j] = f2bf(val);
    }
}

// ---------------- host-side launch helper ----------------
template <typename AT, int NT, int EPI, bool ACC, bool MG>
static void launch_gemm(hipStream_t st,
                        const AT* A, long sAy, long sAz, int lda,
                        const bf16_t* Bt, long sBy, long sBz, int ldb,
                        void* C, long sCy, long sCz, int ldc,
                        int M, int N, int K,
                        const float* bias, const float* resid, int by, int bz) {
    int ntiles = N / (16 * NT);
    int mtiles = (M + 15) / 16;
    int waves = ntiles * mtiles;
    dim3 grid((waves + 7) / 8, by, bz);
    gemm_wmma<AT, NT, EPI, ACC, MG><<<grid, 256, 0, st>>>(
        A, sAy, sAz, lda, Bt, sBy, sBz, ldb, C, sCy, sCz, ldc, M, N, K, bias, resid);
}

static inline long cdivl(long a, long b) { return (a + b - 1) / b; }

extern "C" void kernel_launch(void* const* d_in, const int* in_sizes, int n_in,
                              void* d_out, int out_size, void* d_ws, size_t ws_size,
                              hipStream_t stream) {
    const float* hs   = (const float*)d_in[0];
    const float* gmat = (const float*)d_in[1];
    const float* rele = (const float*)d_in[2];
    const float* Wq = (const float*)d_in[3];  const float* bq = (const float*)d_in[4];
    const float* Wk = (const float*)d_in[5];  const float* bk = (const float*)d_in[6];
    const float* Wv = (const float*)d_in[7];  const float* bv = (const float*)d_in[8];
    const float* Wr = (const float*)d_in[9];  const float* br = (const float*)d_in[10];
    const float* Wo = (const float*)d_in[11]; const float* bo = (const float*)d_in[12];
    const float* g1 = (const float*)d_in[13]; const float* b1 = (const float*)d_in[14];
    const float* Wi = (const float*)d_in[15]; const float* bi = (const float*)d_in[16];
    const float* Wf = (const float*)d_in[17]; const float* bfv = (const float*)d_in[18];
    const float* g2 = (const float*)d_in[19]; const float* b2 = (const float*)d_in[20];
    float* out = (float*)d_out;

    char* ws = (char*)d_ws;
    size_t o = 0;
    auto take = [&](size_t bytes) -> size_t {
        size_t r = o; o = (o + bytes + 255) & ~(size_t)255; return r;
    };
    size_t hsbf_o = take((size_t)BS * HIDN * 2);
    size_t wqT_o  = take((size_t)HIDN * HIDN * 2);
    size_t wkT_o  = take((size_t)HIDN * HIDN * 2);
    size_t wvT_o  = take((size_t)HIDN * HIDN * 2);
    size_t woT_o  = take((size_t)HIDN * HIDN * 2);
    size_t wrT_o  = take((size_t)HD * RELD * 2);
    size_t wiT_o  = take((size_t)FFD * HIDN * 2);
    size_t wfT_o  = take((size_t)HIDN * FFD * 2);
    size_t qf_o   = take((size_t)BS * HIDN * 4);
    size_t kf_o   = take((size_t)BS * HIDN * 4);
    size_t vf_o   = take((size_t)BS * HIDN * 4);
    size_t qb_o   = take((size_t)BS * HIDN * 2);
    size_t kb_o   = take((size_t)BS * HIDN * 2);
    size_t vtb_o  = take((size_t)BS * HIDN * 2);
    size_t sc_o   = take((size_t)BB * NH * SS * SS * 4);      // 50 MB scores f32
    size_t rel_o  = take((size_t)BB * SS * SS * HD * 2);      // 134 MB rel bf16
    // alias post-attention buffers into the rel region (dead after rel-score pass)
    size_t ro = rel_o;
    auto take2 = [&](size_t bytes) -> size_t {
        size_t r = ro; ro = (ro + bytes + 255) & ~(size_t)255; return r;
    };
    size_t pr_o    = take2((size_t)BB * NH * SS * SS * 2);    // probs bf16
    size_t ctxf_o  = take2((size_t)BS * HIDN * 4);
    size_t ctxb_o  = take2((size_t)BS * HIDN * 2);
    size_t opre_o  = take2((size_t)BS * HIDN * 4);
    size_t attnf_o = take2((size_t)BS * HIDN * 4);
    size_t attnb_o = take2((size_t)BS * HIDN * 2);
    size_t ffn1_o  = take2((size_t)BS * FFD * 2);
    size_t fpre_o  = take2((size_t)BS * HIDN * 4);

    bf16_t* hsbf = (bf16_t*)(ws + hsbf_o);
    bf16_t* wqT = (bf16_t*)(ws + wqT_o); bf16_t* wkT = (bf16_t*)(ws + wkT_o);
    bf16_t* wvT = (bf16_t*)(ws + wvT_o); bf16_t* woT = (bf16_t*)(ws + woT_o);
    bf16_t* wrT = (bf16_t*)(ws + wrT_o);
    bf16_t* wiT = (bf16_t*)(ws + wiT_o); bf16_t* wfT = (bf16_t*)(ws + wfT_o);
    float* qf = (float*)(ws + qf_o); float* kf = (float*)(ws + kf_o); float* vf = (float*)(ws + vf_o);
    bf16_t* qb = (bf16_t*)(ws + qb_o); bf16_t* kb = (bf16_t*)(ws + kb_o); bf16_t* vtb = (bf16_t*)(ws + vtb_o);
    float* scores = (float*)(ws + sc_o);
    bf16_t* relbf = (bf16_t*)(ws + rel_o);
    bf16_t* probs = (bf16_t*)(ws + pr_o);
    float* ctxf = (float*)(ws + ctxf_o); bf16_t* ctxb = (bf16_t*)(ws + ctxb_o);
    float* opre = (float*)(ws + opre_o);
    float* attnf = (float*)(ws + attnf_o); bf16_t* attnb = (bf16_t*)(ws + attnb_o);
    bf16_t* ffn1 = (bf16_t*)(ws + ffn1_o);
    float* fpre = (float*)(ws + fpre_o);

    // 1. hidden_states -> bf16
    long nhs = (long)BS * HIDN;
    f2bf_kernel<<<cdivl(nhs, 256), 256, 0, stream>>>(hs, hsbf, nhs);
    // 2. weight transposes to (N x K) bf16
    long nw = (long)HIDN * HIDN;
    transpose_bf_kernel<<<cdivl(nw, 256), 256, 0, stream>>>(Wq, wqT, HIDN, HIDN);
    transpose_bf_kernel<<<cdivl(nw, 256), 256, 0, stream>>>(Wk, wkT, HIDN, HIDN);
    transpose_bf_kernel<<<cdivl(nw, 256), 256, 0, stream>>>(Wv, wvT, HIDN, HIDN);
    transpose_bf_kernel<<<cdivl(nw, 256), 256, 0, stream>>>(Wo, woT, HIDN, HIDN);
    transpose_bf_kernel<<<cdivl((long)HD * RELD, 256), 256, 0, stream>>>(Wr, wrT, RELD, HD);
    transpose_bf_kernel<<<cdivl((long)FFD * HIDN, 256), 256, 0, stream>>>(Wi, wiT, HIDN, FFD);
    transpose_bf_kernel<<<cdivl((long)HIDN * FFD, 256), 256, 0, stream>>>(Wf, wfT, FFD, HIDN);

    // 3. QKV projections: (2048x768) @ (768x768)^T-stored + bias -> f32
    launch_gemm<bf16_t, 4, 0, false, false>(stream, hsbf, 0, 0, HIDN, wqT, 0, 0, HIDN,
                                            qf, 0, 0, HIDN, BS, HIDN, HIDN, bq, nullptr, 1, 1);
    launch_gemm<bf16_t, 4, 0, false, false>(stream, hsbf, 0, 0, HIDN, wkT, 0, 0, HIDN,
                                            kf, 0, 0, HIDN, BS, HIDN, HIDN, bk, nullptr, 1, 1);
    launch_gemm<bf16_t, 4, 0, false, false>(stream, hsbf, 0, 0, HIDN, wvT, 0, 0, HIDN,
                                            vf, 0, 0, HIDN, BS, HIDN, HIDN, bv, nullptr, 1, 1);
    // 4. repack to head layouts (q,k: [b,h,s,d]; v: [b,h,d,s]) as bf16
    pack_qkv_kernel<<<cdivl(nhs, 256), 256, 0, stream>>>(qf, kf, vf, qb, kb, vtb);

    // 5. rel projection: (B*S*S x 64) f32 @ WrT + br -> bf16 [b,i,j,d]  (268 MB stream)
    launch_gemm<float, 4, 1, false, false>(stream, rele, 0, 0, RELD, wrT, 0, 0, RELD,
                                           relbf, 0, 0, HD, BB * SS * SS, HD, RELD, br, nullptr, 1, 1);

    // 6. scores1[b,h,i,j] = q . k        batch y = b*H+h
    launch_gemm<bf16_t, 4, 0, false, false>(stream,
        qb, (long)SS * HD, 0, HD, kb, (long)SS * HD, 0, HD,
        scores, (long)SS * SS, 0, SS, SS, SS, HD, nullptr, nullptr, BB * NH, 1);
    // 7. scores2[b,h,i,j] += q[b,h,i,:] . rel[b,i,j,:]   batch y = i, z = b; rows = h (12 of 16)
    launch_gemm<bf16_t, 4, 0, true, true>(stream,
        qb, (long)HD, (long)NH * SS * HD, SS * HD,
        relbf, (long)SS * HD, (long)SS * SS * HD, HD,
        scores, (long)SS, (long)NH * SS * SS, SS * SS,
        NH, SS, HD, nullptr, nullptr, SS, BB);

    // 8. masked softmax (scale 1/8) -> probs bf16
    softmax_kernel<<<dim3(SS, NH, BB), 256, 0, stream>>>(scores, gmat, probs);

    // 9. ctx[b,h,i,d] = probs @ v   batch y = b*H+h
    launch_gemm<bf16_t, 4, 0, false, false>(stream,
        probs, (long)SS * SS, 0, SS, vtb, (long)HD * SS, 0, SS,
        ctxf, (long)SS * HD, 0, HD, SS, HD, SS, nullptr, nullptr, BB * NH, 1);
    // 10. repack ctx -> [b,s,hid] bf16
    pack_ctx_kernel<<<cdivl(nhs, 256), 256, 0, stream>>>(ctxf, ctxb);

    // 11. output projection + bias + residual(hidden_states)
    launch_gemm<bf16_t, 4, 0, false, false>(stream, ctxb, 0, 0, HIDN, woT, 0, 0, HIDN,
                                            opre, 0, 0, HIDN, BS, HIDN, HIDN, bo, hs, 1, 1);
    // 12. LayerNorm1 -> attn_f (f32) + attn_b (bf16)
    ln_kernel<<<BS, 256, 0, stream>>>(opre, g1, b1, attnf, attnb);

    // 13. FFN up: + bi, exact-erf GELU, -> bf16
    launch_gemm<bf16_t, 4, 2, false, false>(stream, attnb, 0, 0, HIDN, wiT, 0, 0, HIDN,
                                            ffn1, 0, 0, FFD, BS, FFD, HIDN, bi, nullptr, 1, 1);
    // 14. FFN down: + bf + residual(attn_f)
    launch_gemm<bf16_t, 4, 0, false, false>(stream, ffn1, 0, 0, FFD, wfT, 0, 0, FFD,
                                            fpre, 0, 0, HIDN, BS, HIDN, FFD, bfv, attnf, 1, 1);
    // 15. LayerNorm2 -> d_out (f32)
    ln_kernel<<<BS, 256, 0, stream>>>(fpre, g2, b2, out, nullptr);

    (void)in_sizes; (void)n_in; (void)out_size; (void)ws_size;
}
